// Net_85727547228621
// MI455X (gfx1250) — compile-verified, matrix operand order checked by reading, not run on hardware
//
#include <hip/hip_runtime.h>
#include <hip/hip_bf16.h>

// ---------------------------------------------------------------------------
// GIN(3) + MLP head for MI455X (gfx1250, wave32, WMMA + TDM).
// Dense GEMMs use v_wmma_f32_16x16x32_bf16 (bf16 A/B, f32 accumulate).
// GIN-layer weight panels are DMA'd into LDS with tensor_load_to_lds
// (double-buffered, TENSORcnt-synchronized). Edge aggregation: f32 atomics
// into an L2-resident [N,256] accumulator.
// ---------------------------------------------------------------------------

#define HD    256
#define GNUM  512
#define LNEPS 1e-5f

typedef __attribute__((ext_vector_type(8)))  float  v8f;
typedef __attribute__((ext_vector_type(16))) __bf16 v16bf;

union FragU { unsigned u[8]; v16bf b; };

#if __has_builtin(__builtin_amdgcn_tensor_load_to_lds)
#define HAVE_TDM 1
#else
#define HAVE_TDM 0
#endif

#if HAVE_TDM
typedef __attribute__((ext_vector_type(4))) unsigned tdm_v4u;
typedef __attribute__((ext_vector_type(8))) int      tdm_v8i;
typedef __attribute__((ext_vector_type(4))) int      tdm_v4i;

// DMA a [rows x kcols] bf16 tile (row stride kstride elements) into LDS,
// rows packed contiguously (kcols*2 bytes apart). 2D descriptor, groups 2/3
// zero. Issue from ONE wave only; wait with s_wait_tensorcnt.
static __device__ __forceinline__ void tdm_load_panel(
    const unsigned short* gsrc, unsigned lds_byte,
    int rows, int kcols, int kstride) {
  unsigned long long ga = (unsigned long long)(uintptr_t)gsrc;
  tdm_v4u g0;
  g0[0] = 1u;                                   // count=1, load, no gather
  g0[1] = lds_byte;                             // D#.lds_addr
  g0[2] = (unsigned)(ga & 0xFFFFFFFFull);       // global_addr[31:0]
  g0[3] = (unsigned)((ga >> 32) & 0x01FFFFFFull) | (2u << 30);  // [56:32]+type=2
  tdm_v8i g1;
  g1[0] = (int)(1u << 16);                                      // data_size=2B
  g1[1] = (int)(((unsigned)kstride & 0xFFFFu) << 16);           // tensor_dim0 lo
  g1[2] = (int)((((unsigned)kstride >> 16) & 0xFFFFu) |
                (((unsigned)rows & 0xFFFFu) << 16));            // dim0 hi|dim1 lo
  g1[3] = (int)((((unsigned)rows >> 16) & 0xFFFFu) |
                (((unsigned)kcols & 0xFFFFu) << 16));           // dim1 hi|tile_dim0
  g1[4] = (int)((unsigned)rows & 0xFFFFu);                      // tile_dim1
  g1[5] = (int)(unsigned)kstride;                               // dim0_stride lo32
  g1[6] = 0;                                                    // stride hi / dim1_stride lo
  g1[7] = 0;
  tdm_v4i gz = (tdm_v4i)0;
#if defined(__clang_major__) && (__clang_major__ >= 23)
  tdm_v8i g4 = (tdm_v8i)0;
  __builtin_amdgcn_tensor_load_to_lds(g0, g1, gz, gz, g4, 0);
#else
  __builtin_amdgcn_tensor_load_to_lds(g0, g1, gz, gz, 0);
#endif
}
#endif  // HAVE_TDM

static __device__ __forceinline__ unsigned short f2bf(float f) {
  unsigned u = __float_as_uint(f);
  unsigned r = ((u >> 16) & 1u) + 0x7FFFu;   // round-to-nearest-even
  return (unsigned short)((u + r) >> 16);
}

static __device__ __forceinline__ unsigned encf(float f) {
  unsigned u = __float_as_uint(f);
  return (u & 0x80000000u) ? ~u : (u | 0x80000000u);
}
static __device__ __forceinline__ float decf(unsigned u) {
  u = (u & 0x80000000u) ? (u & 0x7FFFFFFFu) : ~u;
  return __uint_as_float(u);
}

// ----------------------------- utility kernels -----------------------------

__global__ __launch_bounds__(256) void zero4_kernel(float4* p, int n4) {
  int i = blockIdx.x * 256 + threadIdx.x;
  if (i < n4) p[i] = make_float4(0.f, 0.f, 0.f, 0.f);
}

// Wt[n*K + k] = bf16( W[(k+row_off)*Nout + n] )   (transpose + convert)
__global__ __launch_bounds__(256) void transpose_bf16_kernel(
    const float* __restrict__ W, unsigned short* __restrict__ Wt,
    int K, int Nout, int row_off) {
  int gid = blockIdx.x * 256 + threadIdx.x;
  if (gid >= K * Nout) return;
  int k = gid % K, n = gid / K;
  Wt[(size_t)n * K + k] = f2bf(W[(size_t)(k + row_off) * Nout + n]);
}

// ------------------------------ edge scatter -------------------------------

__global__ __launch_bounds__(256) void scatter1_kernel(
    const float* __restrict__ V, const int* __restrict__ ei,
    float* __restrict__ AGG1, int E) {
  int e = blockIdx.x * 256 + threadIdx.x;
  if (e >= E) return;
  atomicAdd(&AGG1[ei[E + e]], V[ei[e]]);
}

// one thread = one edge x 4 features (float4 gather, 4 atomic adds)
__global__ __launch_bounds__(256) void scatter_feat_kernel(
    const float* __restrict__ X, const int* __restrict__ ei,
    float* __restrict__ AGG, int E) {
  long long gid = (long long)blockIdx.x * 256 + threadIdx.x;
  if (gid >= (long long)E * 64) return;
  int e  = (int)(gid >> 6);
  int c4 = ((int)gid & 63) << 2;
  int s = ei[e], d = ei[E + e];
  const float4 v = *(const float4*)(X + (size_t)s * HD + c4);
  float* a = AGG + (size_t)d * HD + c4;
  atomicAdd(a + 0, v.x); atomicAdd(a + 1, v.y);
  atomicAdd(a + 2, v.z); atomicAdd(a + 3, v.w);
}

// ------------------------- layer 1 (K=1 outer product) ---------------------

__global__ __launch_bounds__(256) void gin1_ln_relu_kernel(
    const float* __restrict__ V, const float* __restrict__ AGG1,
    const float* __restrict__ W1, const float* __restrict__ b1,
    const float* __restrict__ g, const float* __restrict__ be,
    float* __restrict__ X, int N) {
  int wave = threadIdx.x >> 5, lane = threadIdx.x & 31;
  int row = blockIdx.x * 8 + wave;
  if (row >= N) return;
  float s = V[row] + AGG1[row];
  float h[8]; float sm = 0.f, sq = 0.f;
#pragma unroll
  for (int i = 0; i < 8; ++i) {
    int c = lane + 32 * i;
    h[i] = s * W1[c] + b1[c];
    sm += h[i]; sq += h[i] * h[i];
  }
#pragma unroll
  for (int o = 16; o; o >>= 1) { sm += __shfl_xor(sm, o, 32); sq += __shfl_xor(sq, o, 32); }
  float mu = sm * (1.f / HD);
  float var = sq * (1.f / HD) - mu * mu;
  float rstd = rsqrtf(var + LNEPS);
#pragma unroll
  for (int i = 0; i < 8; ++i) {
    int c = lane + 32 * i;
    float v = (h[i] - mu) * rstd * g[c] + be[c];
    X[(size_t)row * HD + c] = fmaxf(v, 0.f);
  }
}

// --------------------- GIN GEMM (+LayerNorm+ReLU fused) --------------------
// OUT[m,:] = relu(LN( (X[m,:]+AGG[m,:]) @ W + b ))  ;  W pre-transposed bf16.
// Weight B-panels (256 rows x 32 K, 16KB) are TDM-DMA'd into LDS,
// double-buffered; wave 0 issues/waits TENSORcnt, barrier publishes.
#define LDA_G 264  // 256 + 8 pad (ushorts)

__global__ __launch_bounds__(256) void gin_gemm_ln_relu_kernel(
    const float* __restrict__ X, const float* __restrict__ AGG,
    const unsigned short* __restrict__ Wt /*[256][256]*/,
    const float* __restrict__ bias, const float* __restrict__ g,
    const float* __restrict__ be, float* __restrict__ OUT) {
  __shared__ unsigned short aT[16 * LDA_G];
  __shared__ float dT[16 * LDA_G];
#if HAVE_TDM
  __shared__ unsigned short bP[2 * 256 * 32];   // 2 x 16KB weight panels
#endif
  const int tid = threadIdx.x;
  const int m0 = blockIdx.x * 16;
  const int lane = tid & 31, wave = tid >> 5;

#if HAVE_TDM
  const unsigned bP_base = (unsigned)(uintptr_t)(&bP[0]);
  if (wave == 0 && lane == 0)
    tdm_load_panel(Wt, bP_base, 256, 32, 256);          // panel for kt=0
#endif

  // stage A = (X + AGG) as bf16 into LDS
  for (int idx = tid; idx < 16 * 64; idx += 256) {
    int r = idx >> 6, c4 = (idx & 63) << 2;
    const float4 xv = *(const float4*)(X + (size_t)(m0 + r) * HD + c4);
    const float4 av = *(const float4*)(AGG + (size_t)(m0 + r) * HD + c4);
    unsigned short* p = aT + r * LDA_G + c4;
    p[0] = f2bf(xv.x + av.x); p[1] = f2bf(xv.y + av.y);
    p[2] = f2bf(xv.z + av.z); p[3] = f2bf(xv.w + av.w);
  }
  __syncthreads();

  const int nlo = lane & 15;
  const int khi = (lane >> 4) << 3;   // K sub-offset 0/8 (A & B frags)
  const int mlo = (lane >> 4) << 3;   // row offset 0/8 (C/D frag)
  const int n0 = wave * 32;

  v8f c0, c1;
#pragma unroll
  for (int r = 0; r < 8; ++r) { c0[r] = bias[n0 + nlo]; c1[r] = bias[n0 + 16 + nlo]; }

  for (int kt = 0; kt < 8; ++kt) {
    const int kb = kt * 32;
    FragU a, b0, b1;
    const unsigned short* ar = aT + nlo * LDA_G;
#if HAVE_TDM
    if (wave == 0) __builtin_amdgcn_s_wait_tensorcnt(0);  // panel kt landed
    __syncthreads();                                      // publish to all waves
    if (wave == 0 && lane == 0 && kt < 7)                 // prefetch panel kt+1
      tdm_load_panel(Wt + (kt + 1) * 32,
                     bP_base + (unsigned)(((kt + 1) & 1) * 16384), 256, 32, 256);
    const unsigned short* bp = bP + (kt & 1) * 8192;      // ushort offset
    const unsigned short* w0 = bp + (size_t)(n0 + nlo) * 32;
    const unsigned short* w1 = bp + (size_t)(n0 + 16 + nlo) * 32;
#pragma unroll
    for (int j = 0; j < 4; ++j) {
      a.u[j]     = *(const unsigned*)(ar + kb + khi + 2 * j);
      a.u[4 + j] = *(const unsigned*)(ar + kb + 16 + khi + 2 * j);
      b0.u[j]     = *(const unsigned*)(w0 + khi + 2 * j);
      b0.u[4 + j] = *(const unsigned*)(w0 + 16 + khi + 2 * j);
      b1.u[j]     = *(const unsigned*)(w1 + khi + 2 * j);
      b1.u[4 + j] = *(const unsigned*)(w1 + 16 + khi + 2 * j);
    }
#else
    const unsigned short* w0 = Wt + (size_t)(n0 + nlo) * 256;
    const unsigned short* w1 = Wt + (size_t)(n0 + 16 + nlo) * 256;
#pragma unroll
    for (int j = 0; j < 4; ++j) {
      a.u[j]     = *(const unsigned*)(ar + kb + khi + 2 * j);
      a.u[4 + j] = *(const unsigned*)(ar + kb + 16 + khi + 2 * j);
      b0.u[j]     = *(const unsigned*)(w0 + kb + khi + 2 * j);
      b0.u[4 + j] = *(const unsigned*)(w0 + kb + 16 + khi + 2 * j);
      b1.u[j]     = *(const unsigned*)(w1 + kb + khi + 2 * j);
      b1.u[4 + j] = *(const unsigned*)(w1 + kb + 16 + khi + 2 * j);
    }
#endif
    c0 = __builtin_amdgcn_wmma_f32_16x16x32_bf16(false, a.b, false, b0.b, (short)0, c0, false, false);
    c1 = __builtin_amdgcn_wmma_f32_16x16x32_bf16(false, a.b, false, b1.b, (short)0, c1, false, false);
  }

#pragma unroll
  for (int r = 0; r < 8; ++r) {
    dT[(r + mlo) * LDA_G + n0 + nlo]      = c0[r];
    dT[(r + mlo) * LDA_G + n0 + 16 + nlo] = c1[r];
  }
  __syncthreads();

  // fused LayerNorm + ReLU: each wave handles 2 of the 16 rows
#pragma unroll
  for (int rr = 0; rr < 2; ++rr) {
    int r = wave * 2 + rr;
    float h[8]; float sm = 0.f, sq = 0.f;
#pragma unroll
    for (int i = 0; i < 8; ++i) {
      h[i] = dT[r * LDA_G + lane + 32 * i];
      sm += h[i]; sq += h[i] * h[i];
    }
#pragma unroll
    for (int o = 16; o; o >>= 1) { sm += __shfl_xor(sm, o, 32); sq += __shfl_xor(sq, o, 32); }
    float mu = sm * (1.f / HD);
    float var = sq * (1.f / HD) - mu * mu;
    float rstd = rsqrtf(var + LNEPS);
#pragma unroll
    for (int i = 0; i < 8; ++i) {
      int cix = lane + 32 * i;
      float v = (h[i] - mu) * rstd * g[cix] + be[cix];
      OUT[(size_t)(m0 + r) * HD + cix] = fmaxf(v, 0.f);
    }
  }
}

// ------------------------------- FC1 GEMM ----------------------------------
// pre1 = [vertex|x|y|z] @ Wf1 + bf1 ; vertex column folded into C init.
#define LDA1 776  // 768 + 8

__global__ __launch_bounds__(256) void fc1_gemm_kernel(
    const float* __restrict__ V, const float* __restrict__ X,
    const float* __restrict__ Y, const float* __restrict__ Z,
    const unsigned short* __restrict__ Wt /*[512][768]*/,
    const float* __restrict__ W0 /*Wf1 row0, 512*/,
    const float* __restrict__ bias, float* __restrict__ PRE1,
    float* __restrict__ s1, float* __restrict__ q1) {
  __shared__ unsigned short aT[16 * LDA1];
  const int tid = threadIdx.x;
  const int m0 = blockIdx.x * 16;

  for (int idx = tid; idx < 16 * 192; idx += 256) {
    int r = idx / 192, c4 = (idx % 192) * 4;
    const float* src; int c;
    if (c4 < 256)      { src = X; c = c4; }
    else if (c4 < 512) { src = Y; c = c4 - 256; }
    else               { src = Z; c = c4 - 512; }
    float4 v = *(const float4*)(src + (size_t)(m0 + r) * HD + c);
    unsigned short* p = aT + r * LDA1 + c4;
    p[0] = f2bf(v.x); p[1] = f2bf(v.y); p[2] = f2bf(v.z); p[3] = f2bf(v.w);
  }
  __syncthreads();

  const int lane = tid & 31, wave = tid >> 5;
  const int nlo = lane & 15;
  const int khi = (lane >> 4) << 3;
  const int mlo = (lane >> 4) << 3;

  v8f c[4];
#pragma unroll
  for (int t = 0; t < 4; ++t) {
    int n = wave * 64 + t * 16 + nlo;
    float w0n = W0[n], bn = bias[n];
#pragma unroll
    for (int r = 0; r < 8; ++r) c[t][r] = bn + V[m0 + r + mlo] * w0n;
  }

  for (int kb = 0; kb < 768; kb += 32) {
    FragU a;
    const unsigned short* ar = aT + nlo * LDA1;
#pragma unroll
    for (int j = 0; j < 4; ++j) {
      a.u[j]     = *(const unsigned*)(ar + kb + khi + 2 * j);
      a.u[4 + j] = *(const unsigned*)(ar + kb + 16 + khi + 2 * j);
    }
#pragma unroll
    for (int t = 0; t < 4; ++t) {
      FragU b;
      const unsigned short* wr = Wt + (size_t)(wave * 64 + t * 16 + nlo) * 768;
#pragma unroll
      for (int j = 0; j < 4; ++j) {
        b.u[j]     = *(const unsigned*)(wr + kb + khi + 2 * j);
        b.u[4 + j] = *(const unsigned*)(wr + kb + 16 + khi + 2 * j);
      }
      c[t] = __builtin_amdgcn_wmma_f32_16x16x32_bf16(false, a.b, false, b.b, (short)0, c[t], false, false);
    }
  }

#pragma unroll
  for (int t = 0; t < 4; ++t) {
    int n = wave * 64 + t * 16 + nlo;
    float s = 0.f, q = 0.f;
#pragma unroll
    for (int r = 0; r < 8; ++r) {
      float x = c[t][r]; s += x; q += x * x;
      PRE1[(size_t)(m0 + r + mlo) * 512 + n] = x;
    }
    s += __shfl_xor(s, 16, 32); q += __shfl_xor(q, 16, 32);
    if (lane < 16) { atomicAdd(&s1[n], s); atomicAdd(&q1[n], q); }
  }
}

// ------------------------------- FC2 GEMM ----------------------------------
// A = relu(bn1(pre1)) applied on the fly while staging; emits pre2 + bn2 stats
#define LDA2 520  // 512 + 8

__global__ __launch_bounds__(256) void fc2_gemm_kernel(
    const float* __restrict__ PRE1, const float* __restrict__ sc,
    const float* __restrict__ sh,
    const unsigned short* __restrict__ Wt /*[256][512]*/,
    const float* __restrict__ bias, float* __restrict__ PRE2,
    float* __restrict__ s2, float* __restrict__ q2) {
  __shared__ unsigned short aT[16 * LDA2];
  const int tid = threadIdx.x;
  const int m0 = blockIdx.x * 16;

  for (int idx = tid; idx < 16 * 128; idx += 256) {
    int r = idx >> 7, c4 = (idx & 127) << 2;
    float4 v = *(const float4*)(PRE1 + (size_t)(m0 + r) * 512 + c4);
    float4 a = *(const float4*)(sc + c4);
    float4 b = *(const float4*)(sh + c4);
    unsigned short* p = aT + r * LDA2 + c4;
    p[0] = f2bf(fmaxf(v.x * a.x + b.x, 0.f));
    p[1] = f2bf(fmaxf(v.y * a.y + b.y, 0.f));
    p[2] = f2bf(fmaxf(v.z * a.z + b.z, 0.f));
    p[3] = f2bf(fmaxf(v.w * a.w + b.w, 0.f));
  }
  __syncthreads();

  const int lane = tid & 31, wave = tid >> 5;
  const int nlo = lane & 15;
  const int khi = (lane >> 4) << 3;
  const int mlo = (lane >> 4) << 3;
  const int n0 = wave * 32;

  v8f c0, c1;
#pragma unroll
  for (int r = 0; r < 8; ++r) { c0[r] = bias[n0 + nlo]; c1[r] = bias[n0 + 16 + nlo]; }

  for (int kb = 0; kb < 512; kb += 32) {
    FragU a, b0, b1;
    const unsigned short* ar = aT + nlo * LDA2;
    const unsigned short* w0 = Wt + (size_t)(n0 + nlo) * 512;
    const unsigned short* w1 = Wt + (size_t)(n0 + 16 + nlo) * 512;
#pragma unroll
    for (int j = 0; j < 4; ++j) {
      a.u[j]     = *(const unsigned*)(ar + kb + khi + 2 * j);
      a.u[4 + j] = *(const unsigned*)(ar + kb + 16 + khi + 2 * j);
      b0.u[j]     = *(const unsigned*)(w0 + kb + khi + 2 * j);
      b0.u[4 + j] = *(const unsigned*)(w0 + kb + 16 + khi + 2 * j);
      b1.u[j]     = *(const unsigned*)(w1 + kb + khi + 2 * j);
      b1.u[4 + j] = *(const unsigned*)(w1 + kb + 16 + khi + 2 * j);
    }
    c0 = __builtin_amdgcn_wmma_f32_16x16x32_bf16(false, a.b, false, b0.b, (short)0, c0, false, false);
    c1 = __builtin_amdgcn_wmma_f32_16x16x32_bf16(false, a.b, false, b1.b, (short)0, c1, false, false);
  }

#pragma unroll
  for (int t = 0; t < 2; ++t) {
    v8f& cc = t ? c1 : c0;
    int n = n0 + t * 16 + nlo;
    float s = 0.f, q = 0.f;
#pragma unroll
    for (int r = 0; r < 8; ++r) {
      float x = cc[r]; s += x; q += x * x;
      PRE2[(size_t)(m0 + r + mlo) * HD + n] = x;
    }
    s += __shfl_xor(s, 16, 32); q += __shfl_xor(q, 16, 32);
    if (lane < 16) { atomicAdd(&s2[n], s); atomicAdd(&q2[n], q); }
  }
}

// ------------------------------ BN finalize --------------------------------

__global__ __launch_bounds__(256) void bn_finalize_kernel(
    const float* __restrict__ s, const float* __restrict__ q,
    const float* __restrict__ g, const float* __restrict__ b,
    float* __restrict__ sc, float* __restrict__ sh, int C, float invN) {
  int i = blockIdx.x * 256 + threadIdx.x;
  if (i >= C) return;
  float mu = s[i] * invN;
  float var = q[i] * invN - mu * mu;
  float r = rsqrtf(var + LNEPS);
  sc[i] = g[i] * r;
  sh[i] = b[i] - mu * g[i] * r;
}

// --------------------------------- head ------------------------------------

__global__ __launch_bounds__(256) void head1_kernel(
    const float* __restrict__ PRE2, const float* __restrict__ sc,
    const float* __restrict__ sh, const float* __restrict__ Wa,
    const float* __restrict__ ba, const float* __restrict__ Wv,
    const float* __restrict__ bv, const int* __restrict__ batch,
    float* __restrict__ ARAW, float* __restrict__ GSUM,
    float* __restrict__ GCNT, unsigned* __restrict__ AMAX, int N) {
  int wave = threadIdx.x >> 5, lane = threadIdx.x & 31;
  int row = blockIdx.x * 8 + wave;
  if (row >= N) return;
  float aa = 0.f, vv = 0.f;
#pragma unroll
  for (int i = 0; i < 8; ++i) {
    int c = lane + 32 * i;
    float h = fmaxf(PRE2[(size_t)row * HD + c] * sc[c] + sh[c], 0.f);
    aa += h * Wa[c]; vv += h * Wv[c];
  }
#pragma unroll
  for (int o = 16; o; o >>= 1) { aa += __shfl_xor(aa, o, 32); vv += __shfl_xor(vv, o, 32); }
  if (lane == 0) {
    float a = aa + ba[0], v = vv + bv[0];
    ARAW[row] = a;
    atomicMax(AMAX, encf(a));
    int g = batch[row];
    atomicAdd(&GSUM[g], v);
    atomicAdd(&GCNT[g], 1.f);
  }
}

__global__ __launch_bounds__(256) void head2_kernel(
    const float* __restrict__ ARAW, const unsigned* __restrict__ AMAX,
    float* __restrict__ SEXP, int N) {
  __shared__ float red[8];
  int i = blockIdx.x * 256 + threadIdx.x;
  int wave = threadIdx.x >> 5, lane = threadIdx.x & 31;
  float amax = decf(*AMAX);
  float e = (i < N) ? expf(ARAW[i] - amax) : 0.f;
#pragma unroll
  for (int o = 16; o; o >>= 1) e += __shfl_xor(e, o, 32);
  if (lane == 0) red[wave] = e;
  __syncthreads();
  if (threadIdx.x == 0) {
    float t = 0.f;
#pragma unroll
    for (int w = 0; w < 8; ++w) t += red[w];
    atomicAdd(SEXP, t);
  }
}

__global__ __launch_bounds__(256) void head3_kernel(
    const float* __restrict__ ARAW, const unsigned* __restrict__ AMAX,
    const float* __restrict__ SEXP, const float* __restrict__ GSUM,
    const float* __restrict__ GCNT, float* __restrict__ out, int N) {
  int i = blockIdx.x * 256 + threadIdx.x;
  if (i < N) {
    out[i] = ARAW[i] - decf(*AMAX) - logf(*SEXP);
  } else if (i < N + GNUM) {
    int g = i - N;
    float c = fmaxf(GCNT[g], 1.f);
    out[i] = tanhf(GSUM[g] / c);
  }
}

// ------------------------------- launcher ----------------------------------

extern "C" void kernel_launch(void* const* d_in, const int* in_sizes, int n_in,
                              void* d_out, int out_size, void* d_ws, size_t ws_size,
                              hipStream_t stream) {
  (void)n_in; (void)out_size; (void)ws_size;
  const float* V    = (const float*)d_in[0];
  const int*   EI   = (const int*)d_in[1];     // int32 per harness contract
  const int*   BID  = (const int*)d_in[2];
  const float* W1   = (const float*)d_in[3];
  const float* b1   = (const float*)d_in[4];
  const float* W2   = (const float*)d_in[5];
  const float* b2   = (const float*)d_in[6];
  const float* W3   = (const float*)d_in[7];
  const float* b3   = (const float*)d_in[8];
  const float* ln_g = (const float*)d_in[9];
  const float* ln_b = (const float*)d_in[10];
  const float* Wf1  = (const float*)d_in[11];
  const float* bf1  = (const float*)d_in[12];
  const float* Wf2  = (const float*)d_in[13];
  const float* bf2  = (const float*)d_in[14];
  const float* bn1g = (const float*)d_in[15];
  const float* bn1b = (const float*)d_in[16];
  const float* bn2g = (const float*)d_in[17];
  const float* bn2b = (const float*)d_in[18];
  const float* Wa   = (const float*)d_in[19];
  const float* ba   = (const float*)d_in[20];
  const float* Wv   = (const float*)d_in[21];
  const float* bv   = (const float*)d_in[22];
  float* out = (float*)d_out;

  const int N = in_sizes[0];          // 115200
  const int E = in_sizes[1] / 2;      // 921600

  // ------- workspace layout -------
  char* ws = (char*)d_ws;
  size_t o = 0;
  float* AGG  = (float*)(ws + o); o += (size_t)N * HD * 4;   // aliased as PRE2
  float* X    = (float*)(ws + o); o += (size_t)N * HD * 4;
  float* Y    = (float*)(ws + o); o += (size_t)N * HD * 4;
  float* Z    = (float*)(ws + o); o += (size_t)N * HD * 4;
  float* PRE1 = (float*)(ws + o); o += (size_t)N * 512 * 4;
  float* AGG1 = (float*)(ws + o); o += (size_t)N * 4;
  float* ARAW = (float*)(ws + o); o += (size_t)N * 4;
  unsigned short* W2T  = (unsigned short*)(ws + o); o += 256 * 256 * 2;
  unsigned short* W3T  = (unsigned short*)(ws + o); o += 256 * 256 * 2;
  unsigned short* WF1T = (unsigned short*)(ws + o); o += 768 * 512 * 2;
  unsigned short* WF2T = (unsigned short*)(ws + o); o += 512 * 256 * 2;
  float* STATS = (float*)(ws + o);                       // 4104 floats
  float* S1 = STATS,        *Q1 = STATS + 512;
  float* SC1 = STATS + 1024, *SH1 = STATS + 1536;
  float* S2 = STATS + 2048,  *Q2 = STATS + 2304;
  float* SC2 = STATS + 2560, *SH2 = STATS + 2816;
  float* GSUM = STATS + 3072, *GCNT = STATS + 3584;
  unsigned* AMAX = (unsigned*)(STATS + 4096);
  float* SEXP = STATS + 4097;
  float* PRE2 = AGG;  // AGG dead after layer-3 GEMM; reuse for pre2

  const int MB = N / 16;              // GEMM M-blocks (7200)

  // weight prep (tiny): transpose+convert to bf16 [Nout][K]
  transpose_bf16_kernel<<<(256 * 256 + 255) / 256, 256, 0, stream>>>(W2, W2T, 256, 256, 0);
  transpose_bf16_kernel<<<(256 * 256 + 255) / 256, 256, 0, stream>>>(W3, W3T, 256, 256, 0);
  transpose_bf16_kernel<<<(768 * 512 + 255) / 256, 256, 0, stream>>>(Wf1, WF1T, 768, 512, 1);
  transpose_bf16_kernel<<<(512 * 256 + 255) / 256, 256, 0, stream>>>(Wf2, WF2T, 512, 256, 0);

  // zero per-call accumulators
  zero4_kernel<<<(N / 4 + 255) / 256, 256, 0, stream>>>((float4*)AGG1, N / 4);
  zero4_kernel<<<(4104 / 4 + 255) / 256, 256, 0, stream>>>((float4*)STATS, 4104 / 4);

  // ---- layer 1 ----
  scatter1_kernel<<<(E + 255) / 256, 256, 0, stream>>>(V, EI, AGG1, E);
  gin1_ln_relu_kernel<<<N / 8, 256, 0, stream>>>(V, AGG1, W1, b1, ln_g, ln_b, X, N);

  // ---- layer 2 ----
  zero4_kernel<<<((N * HD / 4) + 255) / 256, 256, 0, stream>>>((float4*)AGG, N * HD / 4);
  scatter_feat_kernel<<<(int)(((long long)E * 64 + 255) / 256), 256, 0, stream>>>(X, EI, AGG, E);
  gin_gemm_ln_relu_kernel<<<MB, 256, 0, stream>>>(X, AGG, W2T, b2, ln_g, ln_b, Y);

  // ---- layer 3 ----
  zero4_kernel<<<((N * HD / 4) + 255) / 256, 256, 0, stream>>>((float4*)AGG, N * HD / 4);
  scatter_feat_kernel<<<(int)(((long long)E * 64 + 255) / 256), 256, 0, stream>>>(Y, EI, AGG, E);
  gin_gemm_ln_relu_kernel<<<MB, 256, 0, stream>>>(Y, AGG, W3T, b3, ln_g, ln_b, Z);

  // ---- MLP head ----
  fc1_gemm_kernel<<<MB, 256, 0, stream>>>(V, X, Y, Z, WF1T, Wf1 /*row0*/, bf1, PRE1, S1, Q1);
  bn_finalize_kernel<<<2, 256, 0, stream>>>(S1, Q1, bn1g, bn1b, SC1, SH1, 512, 1.f / (float)N);
  fc2_gemm_kernel<<<MB, 256, 0, stream>>>(PRE1, SC1, SH1, WF2T, bf2, PRE2, S2, Q2);
  bn_finalize_kernel<<<1, 256, 0, stream>>>(S2, Q2, bn2g, bn2b, SC2, SH2, 256, 1.f / (float)N);

  head1_kernel<<<N / 8, 256, 0, stream>>>(PRE2, SC2, SH2, Wa, ba, Wv, bv, BID,
                                          ARAW, GSUM, GCNT, AMAX, N);
  head2_kernel<<<(N + 255) / 256, 256, 0, stream>>>(ARAW, AMAX, SEXP, N);
  head3_kernel<<<(N + GNUM + 255) / 256, 256, 0, stream>>>(ARAW, AMAX, SEXP, GSUM, GCNT, out, N);
}
